// HMM_79353815761527
// MI455X (gfx1250) — matrix operand order repaired and checked by reading, not compiled, not run on hardware
//
#include <hip/hip_runtime.h>
#include <hip/hip_fp16.h>
#include <hip/hip_bf16.h>

// HMM forward filter, chunk-parallel matrix-product scan for gfx1250 (MI455X).
//
// Roofline: outputs are 135MB -> ~6us at 23.3 TB/s, so BW is never the limit.
// The reference scan is a 131072-step serial chain (~ms latency-bound). We
// convert it into 131072 x (128x128x128) f16 WMMA matmuls (~550 GFLOP) via
// per-chunk transfer-matrix products: alpha_t = alpha_{t-1} (Gamma diag(g_t))
// is linear, and since the filter renormalizes every step only the *direction*
// at chunk boundaries matters -> fp16 products with per-step max-renorm are
// legitimate. Outputs are then produced exactly in fp32 by a per-chunk replay.

typedef __attribute__((ext_vector_type(16))) _Float16 v16h;
typedef __attribute__((ext_vector_type(8)))  float    v8f;

#define MST     128
#define DOBS    8
#define TLEN    131072
#define NCHUNK  512
#define CLEN    (TLEN / NCHUNK)       // 256
#define LOG2PI_F 1.8378770664093453f

// ---- workspace layout (bytes) ----
#define WS_GT      0u            // Gamma^T  f32  128*128*4 = 65536
#define WS_GFRAG   65536u        // Gamma, WMMA-B fragment layout, f16, 32768
#define WS_U0      98304u        // u0 f32, 512
#define WS_INVS    98816u        // exp(-log_sigma) f32 128*8*4 = 4096
#define WS_MIS     102912u       // mu*exp(-log_sigma) f32, 4096
#define WS_CTERM   107008u       // per-state log const, 512
#define WS_USTART  107520u       // chunk-start vectors f32 512*128*4 = 262144
#define WS_PMATS   369664u       // chunk products f16 512*128*128*2 = 16777216

// =====================================================================
// Prep: Gamma (row softmax, diag logit pinned 0), Gamma^T, WMMA-B-fragment
// copy of Gamma, u0 softmax, emission constants.
// =====================================================================
__global__ __launch_bounds__(128)
void hmm_prep(const float* __restrict__ tl, const float* __restrict__ il,
              const float* __restrict__ means, const float* __restrict__ ls,
              float* __restrict__ GT, _Float16* __restrict__ Gfrag,
              float* __restrict__ u0, float* __restrict__ invs,
              float* __restrict__ mis, float* __restrict__ cterm)
{
    const int r = threadIdx.x;           // this thread owns Gamma row r
    float row[MST];
    float mx = 0.0f;                     // diagonal logit is 0
    for (int cc = 0; cc < MST; ++cc) {
        float v = (cc == r) ? 0.0f : tl[r*(MST-1) + cc - (cc > r ? 1 : 0)];
        row[cc] = v;
        mx = fmaxf(mx, v);
    }
    float sum = 0.0f;
    for (int cc = 0; cc < MST; ++cc) { float e = __expf(row[cc]-mx); row[cc] = e; sum += e; }
    const float inv = 1.0f / sum;

    // row r of Gamma is K-index r of every B fragment (kb = r>>5).
    const int kb = r >> 5, kk = r & 31, lh = (kk >> 3) & 1, h = (kk & 7) | ((kk >> 4) << 3);
    for (int cc = 0; cc < MST; ++cc) {
        float gv = row[cc] * inv;
        GT[cc*MST + r] = gv;                               // column-major for replay
        int tn = cc >> 4, lane = (lh << 4) | (cc & 15);    // B layout: N=lane&15, K split on lane>>4
        Gfrag[(kb*8 + tn)*512 + lane*16 + h] = (_Float16)gv;
    }

    float ct = -0.5f * (float)DOBS * LOG2PI_F;
    #pragma unroll
    for (int d = 0; d < DOBS; ++d) {
        float l  = ls[r*DOBS + d];
        float iv = __expf(-l);
        invs[r*DOBS + d] = iv;
        mis[r*DOBS + d]  = means[r*DOBS + d] * iv;
        ct -= l;
    }
    cterm[r] = ct;

    if (r == 0) {
        float m0 = il[0];
        for (int i = 1; i < MST; ++i) m0 = fmaxf(m0, il[i]);
        float su = 0.0f;
        for (int i = 0; i < MST; ++i) su += __expf(il[i] - m0);
        float is = 1.0f / su;
        for (int i = 0; i < MST; ++i) u0[i] = __expf(il[i] - m0) * is;
    }
}

// =====================================================================
// Pass 1: per-chunk transfer matrix P_c = prod_t (Gamma * diag(ghat_t)),
// 32 v_wmma_f32_16x16x32_f16 per wave per step, B loads software-pipelined
// one fragment ahead so DS latency overlaps the matrix pipe.
// =====================================================================
__global__ __launch_bounds__(256)
void hmm_chunk_products(const float* __restrict__ y,
                        const _Float16* __restrict__ Gfrag,
                        const float* __restrict__ invs,
                        const float* __restrict__ mis,
                        const float* __restrict__ cterm,
                        _Float16* __restrict__ Pmats)
{
    extern __shared__ char smem[];
    float*    P32  = (float*)smem;                 // 128x128 f32, 64KB (strip-local per wave)
    _Float16* Bf   = (_Float16*)(smem + 65536);    // Gamma B fragments, 32KB
    float*    ghat = (float*)(smem + 98304);       // 128 f32
    float*    redA = ghat + MST;                   // 8 f32 : lg-max partials
    float*    redB = redA + 8;                     // 8 f32 : renorm-max partials

    const int tid  = threadIdx.x;
    const int lane = tid & 31;
    const int wave = tid >> 5;                     // wave w owns P rows [16w,16w+16)
    const int c    = blockIdx.x;

    // stage Gamma fragments (coalesced dwords; hot in L2 across all 512 WGs)
    {
        const uint32_t* src = (const uint32_t*)Gfrag;
        uint32_t*       dst = (uint32_t*)Bf;
        for (int i = tid; i < MST*MST/2; i += 256) dst[i] = src[i];
    }
    // P = I
    for (int i = tid; i < MST*MST; i += 256)
        P32[i] = ((i >> 7) == (i & 127)) ? 1.0f : 0.0f;

    float pin[DOBS], pms[DOBS], pct = 0.0f;
    if (tid < MST) {
        pct = cterm[tid];
        #pragma unroll
        for (int d = 0; d < DOBS; ++d) { pin[d] = invs[tid*DOBS+d]; pms[d] = mis[tid*DOBS+d]; }
    }
    float s = 1.0f;                                // running renorm (uniform => direction-preserving)
    __syncthreads();

    const int m      = lane & 15;
    const int laneHi = lane >> 4;

    auto loadB = [&](int idx) {
        v16h b;
        const _Float16* bp = Bf + idx*512 + lane*16;   // contiguous 32B -> 2x ds_load_b128
        #pragma unroll
        for (int h = 0; h < 16; ++h) b[h] = bp[h];
        return b;
    };

    for (int t = c*CLEN; t < c*CLEN + CLEN; ++t) {
        // ---- emission log-densities -> ghat = exp(lg - max) in (0,1] ----
        float lgv = -3.0e38f;
        if (tid < MST) {
            float a = pct;
            #pragma unroll
            for (int d = 0; d < DOBS; ++d) {
                float z = y[(size_t)t*DOBS + d] * pin[d] - pms[d];
                a -= 0.5f * z * z;
            }
            lgv = a;
        }
        if (t + 1 < TLEN) __builtin_prefetch(&y[(size_t)(t+1)*DOBS], 0, 3);
        float lm = lgv;
        #pragma unroll
        for (int off = 16; off >= 1; off >>= 1)
            lm = fmaxf(lm, __shfl_xor(lm, off, 32));
        if (wave < 4 && lane == 0) redA[wave] = lm;
        __syncthreads();                                             // B1
        const float lgmax = fmaxf(fmaxf(redA[0], redA[1]), fmaxf(redA[2], redA[3]));
        if (tid < MST) ghat[tid] = __expf(lgv - lgmax);
        __syncthreads();                                             // B2

        // ---- A fragments: this wave's 16x128 strip of P (ISA 16-bit A layout),
        //      rescaled by running renorm s and converted to f16 ----
        v16h afr[4];
        {
            const float* Prow = P32 + (wave*16 + m)*MST;
            #pragma unroll
            for (int kbb = 0; kbb < 4; ++kbb) {
                #pragma unroll
                for (int h = 0; h < 16; ++h) {
                    int k = kbb*32 + (h & 7) + ((h >> 3) << 4) + (laneHi << 3);
                    afr[kbb][h] = (_Float16)(Prow[k] * s);
                }
            }
        }

        // ---- P <- (P @ Gamma) * diag(ghat), B loads pipelined 1 frag ahead ----
        float wmax = 0.0f;
        v16h bcur = loadB(0);                       // (kb=0, tn=0)
        #pragma unroll
        for (int tn = 0; tn < 8; ++tn) {
            v8f acc = {};
            #pragma unroll
            for (int kbb = 0; kbb < 4; ++kbb) {
                v16h bnext = bcur;
                if (!(tn == 7 && kbb == 3)) {
                    const int nkb = (kbb == 3) ? 0 : kbb + 1;
                    const int ntn = (kbb == 3) ? tn + 1 : tn;
                    bnext = loadB(nkb*8 + ntn);     // in flight during the wmma below
                }
                acc = __builtin_amdgcn_wmma_f32_16x16x32_f16(false, afr[kbb], false, bcur,
                                                             (short)0, acc, false, false);
                bcur = bnext;
            }
            const float gh = ghat[tn*16 + m];       // C/D layout: N = lane&15 for all 8 elems
            #pragma unroll
            for (int e = 0; e < 8; ++e) {
                float v = acc[e] * gh;              // column scale by diag(ghat)
                wmax = fmaxf(wmax, fabsf(v));
                P32[(wave*16 + laneHi*8 + e)*MST + tn*16 + m] = v;   // M = e + 8*(lane>>4)
            }
        }

        // ---- workgroup max -> next step's rescale ----
        #pragma unroll
        for (int off = 16; off >= 1; off >>= 1)
            wmax = fmaxf(wmax, __shfl_xor(wmax, off, 32));
        if (lane == 0) redB[wave] = wmax;
        __syncthreads();                                             // B3
        float gmax = redB[0];
        #pragma unroll
        for (int w = 1; w < 8; ++w) gmax = fmaxf(gmax, redB[w]);
        gmax = fmaxf(gmax, 1e-30f);
        s = 1.0f / gmax;
    }

    // emit normalized chunk product (f16, max |entry| ~ 1)
    _Float16* Pout = Pmats + (size_t)c * (MST*MST);
    for (int i = tid; i < MST*MST; i += 256)
        Pout[i] = (_Float16)(P32[i] * s);
}

// =====================================================================
// Pass 1.5: serial fold of chunk products into normalized chunk-start
// vectors. 256 threads, split-K partials, wave-shuffle reductions.
// =====================================================================
__global__ __launch_bounds__(256)
void hmm_combine(const _Float16* __restrict__ Pmats,
                 const float* __restrict__ u0,
                 float* __restrict__ ustart)
{
    extern __shared__ char smem[];
    _Float16* Pt   = (_Float16*)smem;             // 32KB chunk-product tile
    float*    u    = (float*)(smem + 32768);      // 128
    float*    redp = u + MST;                     // 256
    float*    red4 = redp + 256;                  // 4

    const int tid  = threadIdx.x;
    const int lane = tid & 31, wave = tid >> 5;
    const int j    = tid & 127;                   // output column
    const int half = tid >> 7;                    // K-split

    if (tid < MST) u[tid] = u0[tid];
    __syncthreads();

    for (int c = 0; c < NCHUNK; ++c) {
        if (tid < MST) ustart[c*MST + tid] = u[tid];   // state entering chunk c
        const uint32_t* src = (const uint32_t*)(Pmats + (size_t)c*MST*MST);
        uint32_t*       dst = (uint32_t*)Pt;
        for (int i = tid; i < MST*MST/2; i += 256) dst[i] = src[i];
        __syncthreads();                                             // Pt visible

        float part = 0.0f;
        const int i0 = half * 64;
        for (int i = 0; i < 64; ++i)
            part += u[i0 + i] * (float)Pt[(i0 + i)*MST + j];
        redp[tid] = part;
        __syncthreads();

        float v = 0.0f, ps = 0.0f;
        if (tid < MST) { v = redp[tid] + redp[tid + 128]; ps = v; }
        #pragma unroll
        for (int off = 16; off >= 1; off >>= 1) ps += __shfl_xor(ps, off, 32);
        if (wave < 4 && lane == 0) red4[wave] = ps;
        __syncthreads();
        const float tot = (red4[0] + red4[1]) + (red4[2] + red4[3]);
        if (tid < MST) u[tid] = v / tot;          // normalized filtered posterior at chunk end
        __syncthreads();
    }
}

// =====================================================================
// Pass 2: exact fp32 replay of the reference recurrence per chunk.
// =====================================================================
__global__ __launch_bounds__(128)
void hmm_replay(const float* __restrict__ y,
                const float* __restrict__ GT,
                const float* __restrict__ invs,
                const float* __restrict__ mis,
                const float* __restrict__ cterm,
                const float* __restrict__ ustart,
                float* __restrict__ outUt,
                float* __restrict__ outft,
                float* __restrict__ outUtt)
{
    extern __shared__ char smem[];
    float* GTl   = (float*)smem;                 // 128x128 padded stride 129 (bank-conflict-free)
    float* uprev = (float*)(smem + 66048);       // 128
    float* red4  = uprev + MST;                  // 4

    const int j    = threadIdx.x;
    const int lane = j & 31, wave = j >> 5;
    const int c    = blockIdx.x;

    for (int idx = j; idx < MST*MST; idx += 128) {
        int col = idx >> 7, rowi = idx & 127;
        GTl[col*129 + rowi] = GT[idx];
    }
    float pin[DOBS], pms[DOBS];
    const float pct = cterm[j];
    #pragma unroll
    for (int d = 0; d < DOBS; ++d) { pin[d] = invs[j*DOBS+d]; pms[d] = mis[j*DOBS+d]; }
    uprev[j] = ustart[c*MST + j];
    __syncthreads();

    const float* gcol = GTl + j*129;
    for (int t = c*CLEN; t < c*CLEN + CLEN; ++t) {
        if (t + 1 < TLEN) __builtin_prefetch(&y[(size_t)(t+1)*DOBS], 0, 3);
        float ut = 0.0f;                          // u_t[j] = (u_prev @ Gamma)[j]
        #pragma unroll 8
        for (int i = 0; i < MST; ++i) ut += uprev[i] * gcol[i];

        float a = pct;                            // log emission, plain exp like the reference
        #pragma unroll
        for (int d = 0; d < DOBS; ++d) {
            float z = y[(size_t)t*DOBS + d] * pin[d] - pms[d];
            a -= 0.5f * z * z;
        }
        const float g = __expf(a);
        const float p = ut * g;

        float ps = p;                             // f_t = sum_j p_j (wave shuffle + 4-way LDS)
        #pragma unroll
        for (int off = 16; off >= 1; off >>= 1) ps += __shfl_xor(ps, off, 32);
        if (lane == 0) red4[wave] = ps;
        __syncthreads();                          // also fences this step's uprev reads
        const float f   = (red4[0] + red4[1]) + (red4[2] + red4[3]);
        const float utt = p / f;
        outUt [(size_t)t*MST + j] = utt;          // filtered
        outUtt[(size_t)t*MST + j] = ut;           // predicted
        if (j == 0) outft[t] = f;
        uprev[j] = utt;
        __syncthreads();
    }
}

// =====================================================================
extern "C" void kernel_launch(void* const* d_in, const int* in_sizes, int n_in,
                              void* d_out, int out_size, void* d_ws, size_t ws_size,
                              hipStream_t stream)
{
    const float* y     = (const float*)d_in[0];   // (T, 8)
    const float* tl    = (const float*)d_in[1];   // (128, 127)
    const float* il    = (const float*)d_in[2];   // (128,)
    const float* means = (const float*)d_in[3];   // (128, 8)
    const float* ls    = (const float*)d_in[4];   // (128, 8)

    char* ws = (char*)d_ws;
    float*    GT     = (float*)   (ws + WS_GT);
    _Float16* Gfrag  = (_Float16*)(ws + WS_GFRAG);
    float*    u0     = (float*)   (ws + WS_U0);
    float*    invs   = (float*)   (ws + WS_INVS);
    float*    mis    = (float*)   (ws + WS_MIS);
    float*    cterm  = (float*)   (ws + WS_CTERM);
    float*    ustart = (float*)   (ws + WS_USTART);
    _Float16* Pmats  = (_Float16*)(ws + WS_PMATS);

    float* out    = (float*)d_out;
    float* outUt  = out;                                  // T*128
    float* outft  = out + (size_t)TLEN * MST;             // T
    float* outUtt = out + (size_t)TLEN * MST + TLEN;      // T*128

    const int smem1 = 98304 + 512 + 32 + 32;              // 98880  (< 320KB/WGP)
    const int smem2 = 32768 + 512 + 1024 + 16;            // 34320
    const int smem3 = 66048 + 512 + 16;                   // 66576

    (void)hipFuncSetAttribute((const void*)hmm_chunk_products,
                              hipFuncAttributeMaxDynamicSharedMemorySize, smem1);
    (void)hipFuncSetAttribute((const void*)hmm_replay,
                              hipFuncAttributeMaxDynamicSharedMemorySize, smem3);

    hmm_prep<<<1, 128, 0, stream>>>(tl, il, means, ls, GT, Gfrag, u0, invs, mis, cterm);
    hmm_chunk_products<<<NCHUNK, 256, smem1, stream>>>(y, Gfrag, invs, mis, cterm, Pmats);
    hmm_combine<<<1, 256, smem2, stream>>>(Pmats, u0, ustart);
    hmm_replay<<<NCHUNK, 128, smem3, stream>>>(y, GT, invs, mis, cterm, ustart,
                                               outUt, outft, outUtt);
}
